// SpectralNetLoss_86483461472335
// MI455X (gfx1250) — matrix operand order repaired and checked by reading, not compiled, not run on hardware
//
#include <hip/hip_runtime.h>

// ---------------------------------------------------------------------------
// loss = (1/(B*N^2)) * sum_b sum_ij W_ij * ( ||Y_j||^2 - <Y_i,Y_j> )
// B=8, N=4096, K=64.  W stays fp32 (read once, 537MB -> ~23us @ 23.3TB/s).
// Gram tiles G = Y_I * Y_J^T computed with v_wmma_f32_16x16x32_f16 (K=64 -> 2 WMMAs).
// ---------------------------------------------------------------------------

typedef __attribute__((ext_vector_type(16))) _Float16 v16h;
typedef __attribute__((ext_vector_type(8)))  _Float16 v8h;
typedef __attribute__((ext_vector_type(8)))  float    v8f;

#define BB 8
#define NN 4096
#define KK 64
#define JSPLIT 8            // j-tiles per wave = (N/16)/JSPLIT = 32

// ---------------------------------------------------------------------------
// Kernel 1: Y (fp32 [B*N,64]) -> Yh (f16) and q = rowwise ||Y||^2 (fp32)
// ---------------------------------------------------------------------------
__global__ void __launch_bounds__(256)
prep_kernel(const float* __restrict__ Y, _Float16* __restrict__ Yh,
            float* __restrict__ q, int rows) {
  int r = blockIdx.x * blockDim.x + threadIdx.x;
  if (r >= rows) return;
  const float4* yp = (const float4*)(Y + (size_t)r * KK);
  _Float16* op = Yh + (size_t)r * KK;
  float s = 0.0f;
#pragma unroll
  for (int i = 0; i < KK / 4; ++i) {
    float4 v = yp[i];
    s += v.x * v.x + v.y * v.y + v.z * v.z + v.w * v.w;
    v8h h;  // build 4 halfs, store 8B at a time via half of v8h? store 4 halfs (8B)
    _Float16 h0 = (_Float16)v.x, h1 = (_Float16)v.y,
             h2 = (_Float16)v.z, h3 = (_Float16)v.w;
    op[4 * i + 0] = h0; op[4 * i + 1] = h1;
    op[4 * i + 2] = h2; op[4 * i + 3] = h3;
    (void)h;
  }
  q[r] = s;
}

// ---------------------------------------------------------------------------
// Fragment loaders (wave32, ISA 7.12.2 layouts), ld = 64 halfs (one Y row).
// base points at (row0, k0) of the f16 matrix.
// ---------------------------------------------------------------------------
// A-matrix 16x32 f16: lane m = l&15; lanes0-15: v0-3=K0..7, v4-7=K16..23;
//                     lanes16-31: v0-3=K8..15, v4-7=K24..31.
__device__ __forceinline__ v16h load_a_frag(const _Float16* base, int lane) {
  const int m = lane & 15, hs = lane >> 4;
  const char* p = (const char*)(base + (size_t)m * KK) + hs * 16;
  v8h lo = *(const v8h*)(p);        // 16 bytes: 8 halfs
  v8h hi = *(const v8h*)(p + 32);   // 16 bytes: 8 halfs
  return __builtin_shufflevector(lo, hi, 0, 1, 2, 3, 4, 5, 6, 7,
                                 8, 9, 10, 11, 12, 13, 14, 15);
}

// B-matrix 32x16 f16: lane n = l&15; lanes0-15: v0-3=K0..7, v4-7=K16..23;
//                     lanes16-31: v0-3=K8..15, v4-7=K24..31.
__device__ __forceinline__ v16h load_b_frag(const _Float16* base, int lane) {
  const int n = lane & 15, hs = lane >> 4;
  const char* p = (const char*)(base + (size_t)n * KK) + hs * 16;
  v8h lo = *(const v8h*)(p);        // halfs K .. K+7
  v8h hi = *(const v8h*)(p + 32);   // halfs K+16 .. K+23
  return __builtin_shufflevector(lo, hi, 0, 1, 2, 3, 4, 5, 6, 7,
                                 8, 9, 10, 11, 12, 13, 14, 15);
}

// ---------------------------------------------------------------------------
// Kernel 2: one wave per (b, i-tile, j-chunk). Streams W once.
// grid.x = B * (N/16) * JSPLIT, block = 32 (one wave).
// ---------------------------------------------------------------------------
__global__ void __launch_bounds__(32)
loss_kernel(const float* __restrict__ W, const _Float16* __restrict__ Yh,
            const float* __restrict__ q, float* __restrict__ partials) {
  const int lane = threadIdx.x & 31;
  const int x = blockIdx.x;
  const int b      = x / ((NN / 16) * JSPLIT);
  const int rem    = x % ((NN / 16) * JSPLIT);
  const int itile  = rem / JSPLIT;
  const int jchunk = rem % JSPLIT;
  const int i0 = itile * 16;

  // A fragments: rows i0..i0+15, K = 0..31 and 32..63 (live for whole j loop)
  const _Float16* yrowA = Yh + ((size_t)b * NN + i0) * KK;
  const v16h a0 = load_a_frag(yrowA, lane);
  const v16h a1 = load_a_frag(yrowA + 32, lane);

  const int rsel = lane & 15;      // C-layout: N (column j within tile)
  const int hs   = lane >> 4;      // C-layout: M = r + 8*hs
  const float* wbase =
      W + ((size_t)b * NN + (size_t)(i0 + 8 * hs)) * NN + rsel;
  const float* qrow = q + (size_t)b * NN;
  const _Float16* yrowB = Yh + (size_t)b * NN * KK;

  float acc = 0.0f;

  const int jt0 = jchunk * (NN / 16 / JSPLIT);   // first j-tile (32 per wave)
#pragma unroll 1
  for (int t = 0; t < NN / 16 / JSPLIT; ++t) {
    const int j0 = (jt0 + t) * 16;

    // Prefetch next W tile column block (gfx1250 global_prefetch_b8)
    __builtin_prefetch(wbase + (size_t)(j0 + 16), 0, 1);

    // B fragments: rows j0..j0+15 of Yh, transposed role (B[k][n] = Yh[j0+n][k])
    const _Float16* yb = yrowB + (size_t)j0 * KK;
    v16h b0 = load_b_frag(yb, lane);
    v16h b1 = load_b_frag(yb + 32, lane);

    // G = Y_I * Y_J^T  (fp32 accumulate)
    v8f c = {};
    c = __builtin_amdgcn_wmma_f32_16x16x32_f16(false, a0, false, b0,
                                               (short)0, c, false, false);
    c = __builtin_amdgcn_wmma_f32_16x16x32_f16(false, a1, false, b1,
                                               (short)0, c, false, false);

    // W tile in C-matrix lane layout: lane -> col j0+rsel, vgpr r -> row +r
    const float* wp = wbase + j0;
    float w[8];
#pragma unroll
    for (int r = 0; r < 8; ++r) w[r] = wp[(size_t)r * NN];

    const float qv = qrow[j0 + rsel];   // ||Y_j||^2, j = j0 + (lane&15)

#pragma unroll
    for (int r = 0; r < 8; ++r) acc = fmaf(w[r], qv - c[r], acc);
  }

  // wave32 reduction (deterministic tree)
#pragma unroll
  for (int o = 16; o > 0; o >>= 1) acc += __shfl_xor(acc, o, 32);
  if (lane == 0) partials[blockIdx.x] = acc;
}

// ---------------------------------------------------------------------------
// Kernel 3: deterministic fixed-order reduction of wave partials -> scalar
// ---------------------------------------------------------------------------
__global__ void __launch_bounds__(256)
reduce_kernel(const float* __restrict__ partials, int n, float* __restrict__ out,
              float scale) {
  __shared__ float sm[256];
  float s = 0.0f;
  for (int i = threadIdx.x; i < n; i += 256) s += partials[i];
  sm[threadIdx.x] = s;
  __syncthreads();
#pragma unroll
  for (int o = 128; o > 0; o >>= 1) {
    if ((int)threadIdx.x < o) sm[threadIdx.x] += sm[threadIdx.x + o];
    __syncthreads();
  }
  if (threadIdx.x == 0) out[0] = sm[0] * scale;
}

// ---------------------------------------------------------------------------
extern "C" void kernel_launch(void* const* d_in, const int* in_sizes, int n_in,
                              void* d_out, int out_size, void* d_ws, size_t ws_size,
                              hipStream_t stream) {
  const float* W = (const float*)d_in[0];   // [B, N, N] fp32
  const float* Y = (const float*)d_in[1];   // [B, N, K] fp32
  float* out = (float*)d_out;               // scalar fp32

  // Workspace layout: [Yh f16: B*N*K][q f32: B*N][partials f32: nblocks]
  _Float16* Yh = (_Float16*)d_ws;
  float* q = (float*)((char*)d_ws + (size_t)BB * NN * KK * sizeof(_Float16));
  float* partials = q + (size_t)BB * NN;

  const int rows = BB * NN;                          // 32768
  prep_kernel<<<rows / 256, 256, 0, stream>>>(Y, Yh, q, rows);

  const int nblocks = BB * (NN / 16) * JSPLIT;       // 16384 waves
  loss_kernel<<<nblocks, 32, 0, stream>>>(W, Yh, q, partials);

  // 1/(B*N*N) = 2^-27 exactly
  reduce_kernel<<<1, 256, 0, stream>>>(partials, nblocks, out,
                                       1.0f / ((float)BB * (float)NN * (float)NN));
}
